// downprompt_62216896250009
// MI455X (gfx1250) — compile-verified, matrix operand order checked by reading, not run on hardware
//
#include <hip/hip_runtime.h>
#include <hip/hip_bf16.h>

// ---------------------------------------------------------------------------
// MI455X (gfx1250) DownPrompt GNN pipeline.
//  - Dense GEMMs on v_wmma_f32_16x16x32_bf16 (wave32, fp32 accum).
//    64x64 block tile, 8 waves (4Mx2N), each wave 16x32 (2 accumulators,
//    A-fragment reused), K chunk 64, LDS double-buffered.
//  - Weights pre-transposed/converted to bf16 [Nc][K]; B tiles streamed to
//    LDS with global_load_async_to_lds_b128 (ASYNCcnt) -- no VGPR roundtrip.
//  - Graph conv gather/scatter is HBM-bound -> atomicAdd scatter kernels.
//  - concat([n, mean[batch]]) @ Win split into n@Wtop + (mean@Wbot)[batch].
// ---------------------------------------------------------------------------

typedef __attribute__((ext_vector_type(16))) __bf16 v16bf;
typedef __attribute__((ext_vector_type(8)))  float  v8f;

#define D_DIM 128
#define H_DIM 256
#define G_NUM 500
#define C_NUM 10
#define P_NUM 5

__device__ __forceinline__ unsigned short f2bf(float f) {
    unsigned int u = __float_as_uint(f);
    unsigned int r = (u + 0x7FFFu + ((u >> 16) & 1u)) >> 16;   // RNE
    return (unsigned short)r;
}

// ----------------------------- WMMA GEMM -----------------------------------
// C[M x Nc] = A[M x K] @ B[K x Nc] (+ bias); A fp32 row-major,
// Bt = bf16 pre-transposed [Nc x K].
#define BM 64
#define BN 64
#define KC 64

__global__ __launch_bounds__(256)
void wmma_gemm_bf16(const float* __restrict__ A, const unsigned short* __restrict__ Bt,
                    const float* __restrict__ bias, float* __restrict__ C,
                    int M, int K, int Nc) {
    __shared__ __align__(16) unsigned short As[2][BM * KC];  // [row][k]
    __shared__ __align__(16) unsigned short Bs[2][BN * KC];  // [n][k]

    const int tid  = threadIdx.x;
    const int lane = tid & 31;
    const int wave = tid >> 5;
    const int wm   = wave >> 1;              // 0..3  (M sub-tile)
    const int wn   = wave & 1;               // 0..1  (N sub-tile pair)
    const int l16  = lane & 15;
    const int hl   = lane >> 4;              // lane half per ISA 16-bit layout
    const int tileM = blockIdx.y * BM;
    const int tileN = blockIdx.x * BN;

    // stage K-chunk k0 into buffer p: A via VGPR convert, B via async->LDS
    auto stage = [&](int p, int k0) {
        // B tile: 64 rows x 64 bf16 (128B/row) = 512 x 16B chunks, async copy
        #pragma unroll
        for (int it = 0; it < 2; ++it) {
            int idx = tid + it * 256;        // 0..511
            int n   = idx >> 3;              // 0..63
            int ch  = idx & 7;               // 0..7 (16B chunks)
            const unsigned short* g = Bt + (size_t)(tileN + n) * K + k0 + ch * 8;
            unsigned lds = (unsigned)(size_t)(&Bs[p][n * KC + ch * 8]);
            asm volatile("global_load_async_to_lds_b128 %0, %1, off"
                         :: "v"(lds), "v"((unsigned long long)(size_t)g)
                         : "memory");
        }
        // A tile: 64 rows x 64 f32, convert to bf16 through VGPRs
        #pragma unroll
        for (int it = 0; it < 4; ++it) {
            int idx = tid + it * 256;        // 0..1023
            int r   = idx >> 4;              // 0..63
            int cg  = (idx & 15) * 4;        // 0..60
            int gr  = tileM + r;
            int grc = gr < M ? gr : (M - 1);
            const float* src = A + (size_t)grc * K + k0 + cg;
            float4 f = *(const float4*)src;
            if (k0 + KC < K)
                __builtin_prefetch(src + KC, 0, 0);
            unsigned short* dst = &As[p][r * KC + cg];
            dst[0] = f2bf(f.x); dst[1] = f2bf(f.y);
            dst[2] = f2bf(f.z); dst[3] = f2bf(f.w);
        }
    };

    v8f acc0 = {}, acc1 = {};
    int parity = 0;
    stage(0, 0);
    asm volatile("s_wait_asynccnt 0x0" ::: "memory");
    __syncthreads();

    for (int k0 = 0; k0 < K; k0 += KC) {
        if (k0 + KC < K) stage(parity ^ 1, k0 + KC);

        const unsigned short* arow  = &As[parity][(wm * 16 + l16) * KC];
        const unsigned short* brow0 = &Bs[parity][(wn * 32 + l16) * KC];
        const unsigned short* brow1 = &Bs[parity][(wn * 32 + 16 + l16) * KC];
        #pragma unroll
        for (int h2 = 0; h2 < KC; h2 += 32) {
            v16bf afrag, bfrag0, bfrag1;
            uint4* ap = reinterpret_cast<uint4*>(&afrag);
            ap[0] = *reinterpret_cast<const uint4*>(arow + h2 + hl * 8);
            ap[1] = *reinterpret_cast<const uint4*>(arow + h2 + 16 + hl * 8);
            uint4* b0p = reinterpret_cast<uint4*>(&bfrag0);
            b0p[0] = *reinterpret_cast<const uint4*>(brow0 + h2 + hl * 8);
            b0p[1] = *reinterpret_cast<const uint4*>(brow0 + h2 + 16 + hl * 8);
            uint4* b1p = reinterpret_cast<uint4*>(&bfrag1);
            b1p[0] = *reinterpret_cast<const uint4*>(brow1 + h2 + hl * 8);
            b1p[1] = *reinterpret_cast<const uint4*>(brow1 + h2 + 16 + hl * 8);
            acc0 = __builtin_amdgcn_wmma_f32_16x16x32_bf16(
                       false, afrag, false, bfrag0, (short)0, acc0, false, false);
            acc1 = __builtin_amdgcn_wmma_f32_16x16x32_bf16(
                       false, afrag, false, bfrag1, (short)0, acc1, false, false);
        }
        if (k0 + KC < K)
            asm volatile("s_wait_asynccnt 0x0" ::: "memory");
        __syncthreads();
        parity ^= 1;
    }

    // epilogue per C/D layout: row = base + hl*8 + r, col = l16
    const int col0 = tileN + wn * 32 + l16;
    const int col1 = col0 + 16;
    const float bv0 = bias ? bias[col0] : 0.0f;
    const float bv1 = bias ? bias[col1] : 0.0f;
    #pragma unroll
    for (int r = 0; r < 8; ++r) {
        int row = tileM + wm * 16 + hl * 8 + r;
        if (row < M) {
            C[(size_t)row * Nc + col0] = acc0[r] + bv0;
            C[(size_t)row * Nc + col1] = acc1[r] + bv1;
        }
    }
}

// W[K x Nc] fp32 -> Wt[Nc x K] bf16
__global__ void transpose_to_bf16(const float* __restrict__ src,
                                  unsigned short* __restrict__ dst, int K, int Nc) {
    int i = blockIdx.x * blockDim.x + threadIdx.x;
    if (i >= K * Nc) return;
    int n = i / K, k = i - n * K;
    dst[i] = f2bf(src[(size_t)k * Nc + n]);
}

// --------------------------- elementwise / graph ----------------------------

__global__ void fill_kernel(float* __restrict__ p, float v, int n) {
    int i = blockIdx.x * blockDim.x + threadIdx.x;
    if (i < n) p[i] = v;
}

__global__ void deg_edges(const int* __restrict__ dst, float* __restrict__ deg, int E) {
    int e = blockIdx.x * blockDim.x + threadIdx.x;
    if (e < E) atomicAdd(&deg[dst[e]], 1.0f);
}

__global__ void deg_to_dinv(float* __restrict__ dinv, int N) {
    int i = blockIdx.x * blockDim.x + threadIdx.x;
    if (i < N) dinv[i] = rsqrtf(fmaxf(dinv[i], 1.0f));
}

__global__ void count_nodes(const int* __restrict__ batch, float* __restrict__ cnt, int N) {
    int i = blockIdx.x * blockDim.x + threadIdx.x;
    if (i < N) atomicAdd(&cnt[batch[i]], 1.0f);
}

__global__ void sigmoid2(const float* __restrict__ pa, const float* __restrict__ pb,
                         float* __restrict__ ab) {
    if (threadIdx.x == 0) ab[0] = 1.0f / (1.0f + expf(-pa[0]));
    if (threadIdx.x == 1) ab[1] = 1.0f / (1.0f + expf(-pb[0]));
}

__global__ void conv_init(const float* __restrict__ hw, const float* __restrict__ dinv,
                          const float* __restrict__ bias, const float* __restrict__ res,
                          float* __restrict__ out, int total) {
    int i = blockIdx.x * blockDim.x + threadIdx.x;
    if (i >= total) return;
    int node = i >> 7, d = i & 127;
    float dv = dinv[node];
    float v = bias[d] + hw[i] * dv * dv;
    if (res) v += res[i];
    out[i] = v;
}

__global__ void conv_scatter(const float* __restrict__ hw, const float* __restrict__ dinv,
                             const int* __restrict__ src, const int* __restrict__ dst,
                             float* __restrict__ out, int E) {
    int t = blockIdx.x * blockDim.x + threadIdx.x;
    if (t >= E * 32) return;
    int e  = t >> 5;
    int fg = (t & 31) * 4;
    int s = src[e], d = dst[e];
    float w = dinv[s] * dinv[d];
    const float* hs = hw + (size_t)s * D_DIM + fg;
    float* od = out + (size_t)d * D_DIM + fg;
    atomicAdd(&od[0], hs[0] * w);
    atomicAdd(&od[1], hs[1] * w);
    atomicAdd(&od[2], hs[2] * w);
    atomicAdd(&od[3], hs[3] * w);
}

__global__ void seg_sum(const float* __restrict__ v, const int* __restrict__ batch,
                        float* __restrict__ gsum, int N) {
    int t = blockIdx.x * blockDim.x + threadIdx.x;
    if (t >= N * 32) return;
    int i = t >> 5, fg = (t & 31) * 4;
    int g = batch[i];
    const float* p = v + (size_t)i * D_DIM + fg;
    float* o = gsum + (size_t)g * D_DIM + fg;
    atomicAdd(&o[0], p[0]); atomicAdd(&o[1], p[1]);
    atomicAdd(&o[2], p[2]); atomicAdd(&o[3], p[3]);
}

__global__ void seg_div(const float* __restrict__ gsum, const float* __restrict__ cnt,
                        float* __restrict__ m, int total) {
    int i = blockIdx.x * blockDim.x + threadIdx.x;
    if (i >= total) return;
    m[i] = gsum[i] / fmaxf(cnt[i >> 7], 1.0f);
}

__global__ void combine3(const float* __restrict__ x0, const float* __restrict__ x1,
                         const float* __restrict__ x2, const float* __restrict__ ab,
                         float* __restrict__ out, int n) {
    int i = blockIdx.x * blockDim.x + threadIdx.x;
    if (i < n) out[i] = ab[0] * x0[i] + ab[1] * x1[i] + x2[i];
}

__global__ void add_gather(float* __restrict__ u, const float* __restrict__ q,
                           const int* __restrict__ batch, int N) {
    int t = blockIdx.x * blockDim.x + threadIdx.x;
    if (t >= N * H_DIM) return;
    int i = t >> 8, h = t & 255;
    u[t] += q[(size_t)batch[i] * H_DIM + h];
}

__global__ void mul_ew(const float* __restrict__ a, const float* __restrict__ b,
                       float* __restrict__ out, int n) {
    int i = blockIdx.x * blockDim.x + threadIdx.x;
    if (i < n) out[i] = a[i] * b[i];
}

// ----------------------------- GPFplusAtt ----------------------------------

__global__ void att_weights(const float* __restrict__ embed, const float* __restrict__ aW,
                            const float* __restrict__ ab, float* __restrict__ wbuf, int N) {
    int i = blockIdx.x * blockDim.x + threadIdx.x;
    if (i >= N) return;
    float s[P_NUM];
    #pragma unroll
    for (int p = 0; p < P_NUM; ++p) s[p] = ab[p];
    const float* row = embed + (size_t)i * D_DIM;
    for (int d = 0; d < D_DIM; ++d) {
        float xv = row[d];
        #pragma unroll
        for (int p = 0; p < P_NUM; ++p) s[p] += xv * aW[d * P_NUM + p];
    }
    float mx = s[0];
    #pragma unroll
    for (int p = 1; p < P_NUM; ++p) mx = fmaxf(mx, s[p]);
    float sum = 0.0f;
    #pragma unroll
    for (int p = 0; p < P_NUM; ++p) { s[p] = expf(s[p] - mx); sum += s[p]; }
    #pragma unroll
    for (int p = 0; p < P_NUM; ++p) wbuf[(size_t)i * P_NUM + p] = s[p] / sum;
}

__global__ void prompt_add(float* __restrict__ embed, const float* __restrict__ wbuf,
                           const float* __restrict__ plist, int total) {
    int t = blockIdx.x * blockDim.x + threadIdx.x;
    if (t >= total) return;
    int i = t >> 7, d = t & 127;
    float s = 0.0f;
    #pragma unroll
    for (int p = 0; p < P_NUM; ++p)
        s += wbuf[(size_t)i * P_NUM + p] * plist[p * D_DIM + d];
    embed[t] += s;
}

// ------------------------------- readout -----------------------------------

__global__ void gather_rows(const float* __restrict__ gm, const int* __restrict__ idx,
                            float* __restrict__ rawret, int total) {
    int t = blockIdx.x * blockDim.x + threadIdx.x;
    if (t >= total) return;
    int j = t >> 7, d = t & 127;
    rawret[t] = gm[(size_t)idx[j] * D_DIM + d];
}

__global__ void label_count(const int* __restrict__ labels, float* __restrict__ ccnt, int NI) {
    int j = blockIdx.x * blockDim.x + threadIdx.x;
    if (j < NI) atomicAdd(&ccnt[labels[j]], 1.0f);
}

__global__ void class_sum(const float* __restrict__ rawret, const int* __restrict__ labels,
                          float* __restrict__ csum, int total) {
    int t = blockIdx.x * blockDim.x + threadIdx.x;
    if (t >= total) return;
    int j = t >> 7, d = t & 127;
    atomicAdd(&csum[(size_t)labels[j] * D_DIM + d], rawret[t]);
}

__global__ void class_norm(const float* __restrict__ csum, const float* __restrict__ ccnt,
                           float* __restrict__ an) {
    int c = blockIdx.x, d = threadIdx.x;     // 128 threads
    __shared__ float red[D_DIM];
    float ave = csum[c * D_DIM + d] / fmaxf(ccnt[c], 1.0f);
    red[d] = ave * ave;
    __syncthreads();
    for (int s = 64; s > 0; s >>= 1) {
        if (d < s) red[d] += red[d + s];
        __syncthreads();
    }
    an[c * D_DIM + d] = ave / fmaxf(sqrtf(red[0]), 1e-12f);
}

__global__ void final_softmax(const float* __restrict__ rawret, const float* __restrict__ an,
                              float* __restrict__ out) {
    int j = blockIdx.x, d = threadIdx.x;     // 128 threads
    __shared__ float red[D_DIM];
    __shared__ float dots[C_NUM];
    float r = rawret[(size_t)j * D_DIM + d];
    red[d] = r * r;
    __syncthreads();
    for (int s = 64; s > 0; s >>= 1) {
        if (d < s) red[d] += red[d + s];
        __syncthreads();
    }
    float rn = r / fmaxf(sqrtf(red[0]), 1e-12f);
    for (int c = 0; c < C_NUM; ++c) {
        __syncthreads();
        red[d] = rn * an[c * D_DIM + d];
        __syncthreads();
        for (int s = 64; s > 0; s >>= 1) {
            if (d < s) red[d] += red[d + s];
            __syncthreads();
        }
        if (d == 0) dots[c] = red[0];
    }
    __syncthreads();
    if (d == 0) {
        float mx = dots[0];
        for (int c = 1; c < C_NUM; ++c) mx = fmaxf(mx, dots[c]);
        float e[C_NUM], sum = 0.0f;
        for (int c = 0; c < C_NUM; ++c) { e[c] = expf(dots[c] - mx); sum += e[c]; }
        for (int c = 0; c < C_NUM; ++c) out[j * C_NUM + c] = e[c] / sum;
    }
}

// ------------------------------- driver -------------------------------------

static inline dim3 g1d(long long n) { return dim3((unsigned)((n + 255) / 256)); }

extern "C" void kernel_launch(void* const* d_in, const int* in_sizes, int n_in,
                              void* d_out, int out_size, void* d_ws, size_t ws_size,
                              hipStream_t stream) {
    const float* x      = (const float*)d_in[0];
    const int*   eidx   = (const int*)d_in[1];
    const int*   batch  = (const int*)d_in[2];
    const int*   idx    = (const int*)d_in[3];
    const int*   labels = (const int*)d_in[4];
    const float* W0 = (const float*)d_in[5];  const float* b0 = (const float*)d_in[6];
    const float* W1 = (const float*)d_in[7];  const float* b1 = (const float*)d_in[8];
    const float* W2 = (const float*)d_in[9];  const float* b2 = (const float*)d_in[10];
    const float* cinW = (const float*)d_in[11]; const float* cinB = (const float*)d_in[12];
    const float* chW  = (const float*)d_in[13]; const float* chB  = (const float*)d_in[14];
    const float* coW  = (const float*)d_in[15]; const float* coB  = (const float*)d_in[16];
    const float* plist = (const float*)d_in[17];
    const float* aW    = (const float*)d_in[18];
    const float* aB    = (const float*)d_in[19];
    const float* pa    = (const float*)d_in[20];
    const float* pb    = (const float*)d_in[21];
    float* out = (float*)d_out;

    const int N  = in_sizes[0] / D_DIM;
    const int E  = in_sizes[1] / 2;
    const int NI = in_sizes[3];
    const int TH = in_sizes[11] / (2 * D_DIM * H_DIM);
    const int* esrc = eidx;
    const int* edst = eidx + E;

    // workspace carve (floats then bf16 weights; 16B-aligned slots)
    float* W = (float*)d_ws;
    size_t off = 0;
    auto alloc = [&](size_t n) { float* p = W + off; off += (n + 3) & ~(size_t)3; return p; };
    float* dinv  = alloc(N);
    float* cnt   = alloc(G_NUM);
    float* ab    = alloc(2);
    float* ccnt  = alloc(C_NUM);
    float* xcur  = alloc((size_t)N * D_DIM);
    float* hw    = alloc((size_t)N * D_DIM);
    float* n0    = alloc((size_t)N * D_DIM);
    float* n1    = alloc((size_t)N * D_DIM);
    float* n2    = alloc((size_t)N * D_DIM);
    float* tmp   = alloc((size_t)N * D_DIM);
    float* u     = alloc((size_t)N * H_DIM);
    float* v     = alloc((size_t)N * H_DIM);
    float* m0    = alloc((size_t)G_NUM * D_DIM);
    float* m1    = alloc((size_t)G_NUM * D_DIM);
    float* m2    = alloc((size_t)G_NUM * D_DIM);
    float* gmean = alloc((size_t)G_NUM * D_DIM);
    float* q     = alloc((size_t)G_NUM * H_DIM);
    float* gsum  = alloc((size_t)G_NUM * D_DIM);
    float* wbuf  = alloc((size_t)N * P_NUM);
    float* raw   = alloc((size_t)NI * D_DIM);
    float* csum  = alloc((size_t)C_NUM * D_DIM);
    float* an    = alloc((size_t)C_NUM * D_DIM);
    unsigned short* US = (unsigned short*)(W + off);
    size_t uoff = 0;
    auto ualloc = [&](size_t n) { unsigned short* p = US + uoff;
                                  uoff += (n + 7) & ~(size_t)7; return p; };
    unsigned short* w0t = ualloc(D_DIM * D_DIM);
    unsigned short* w1t = ualloc(D_DIM * D_DIM);
    unsigned short* w2t = ualloc(D_DIM * D_DIM);
    unsigned short* wtT = ualloc((size_t)TH * H_DIM * D_DIM);   // Wtop^T per layer
    unsigned short* wbT = ualloc((size_t)TH * H_DIM * D_DIM);   // Wbot^T per layer
    unsigned short* chT = ualloc((size_t)TH * H_DIM * H_DIM);
    unsigned short* coT = ualloc((size_t)TH * D_DIM * H_DIM);
    (void)ws_size; (void)n_in; (void)out_size;

    auto gemm = [&](const float* A, const unsigned short* Bt, const float* bias, float* C,
                    int M, int K, int Nc) {
        dim3 grid((Nc + BN - 1) / BN, (M + BM - 1) / BM);
        wmma_gemm_bf16<<<grid, 256, 0, stream>>>(A, Bt, bias, C, M, K, Nc);
    };
    auto conv = [&](const float* h, const unsigned short* Wt, const float* bc,
                    const float* res, float* o) {
        gemm(h, Wt, nullptr, hw, N, D_DIM, D_DIM);
        conv_init<<<g1d((long long)N * D_DIM), 256, 0, stream>>>(hw, dinv, bc, res, o, N * D_DIM);
        conv_scatter<<<g1d((long long)E * 32), 256, 0, stream>>>(hw, dinv, esrc, edst, o, E);
    };
    auto segmean = [&](const float* vin, float* m) {
        fill_kernel<<<g1d(G_NUM * D_DIM), 256, 0, stream>>>(gsum, 0.0f, G_NUM * D_DIM);
        seg_sum<<<g1d((long long)N * 32), 256, 0, stream>>>(vin, batch, gsum, N);
        seg_div<<<g1d(G_NUM * D_DIM), 256, 0, stream>>>(gsum, cnt, m, G_NUM * D_DIM);
    };
    auto xpose = [&](const float* src, unsigned short* dst, int K, int Nc) {
        transpose_to_bf16<<<g1d((long long)K * Nc), 256, 0, stream>>>(src, dst, K, Nc);
    };

    // --- one-time weight transpose/convert to bf16 [Nc][K] ---
    xpose(W0, w0t, D_DIM, D_DIM);
    xpose(W1, w1t, D_DIM, D_DIM);
    xpose(W2, w2t, D_DIM, D_DIM);
    for (int i = 0; i < TH; ++i) {
        const float* Wtop = cinW + (size_t)i * 2 * D_DIM * H_DIM;
        xpose(Wtop,                           wtT + (size_t)i * H_DIM * D_DIM, D_DIM, H_DIM);
        xpose(Wtop + (size_t)D_DIM * H_DIM,   wbT + (size_t)i * H_DIM * D_DIM, D_DIM, H_DIM);
        xpose(chW + (size_t)i * H_DIM * H_DIM, chT + (size_t)i * H_DIM * H_DIM, H_DIM, H_DIM);
        xpose(coW + (size_t)i * H_DIM * D_DIM, coT + (size_t)i * D_DIM * H_DIM, H_DIM, D_DIM);
    }

    // --- setup: degrees, counts, mixing scalars, x copy ---
    fill_kernel<<<g1d(N), 256, 0, stream>>>(dinv, 1.0f, N);           // self loop
    deg_edges<<<g1d(E), 256, 0, stream>>>(edst, dinv, E);
    deg_to_dinv<<<g1d(N), 256, 0, stream>>>(dinv, N);
    fill_kernel<<<g1d(G_NUM), 256, 0, stream>>>(cnt, 0.0f, G_NUM);
    count_nodes<<<g1d(N), 256, 0, stream>>>(batch, cnt, N);
    sigmoid2<<<1, 32, 0, stream>>>(pa, pb, ab);
    hipMemcpyAsync(xcur, x, (size_t)N * D_DIM * sizeof(float),
                   hipMemcpyDeviceToDevice, stream);

    // --- think layers ---
    for (int i = 0; i < TH; ++i) {
        conv(xcur, w0t, b0, nullptr, n0);  segmean(n0, m0);
        conv(n0,   w1t, b1, n0,      n1);  segmean(n1, m1);
        conv(n1,   w2t, b2, n1,      n2);  segmean(n2, m2);
        combine3<<<g1d((long long)N * D_DIM), 256, 0, stream>>>(n0, n1, n2, ab, tmp, N * D_DIM);
        combine3<<<g1d(G_NUM * D_DIM), 256, 0, stream>>>(m0, m1, m2, ab, gmean, G_NUM * D_DIM);
        // cond_net: [tmp, gmean[batch]] @ Win = tmp@Wtop + (gmean@Wbot)[batch]
        gemm(gmean, wbT + (size_t)i * H_DIM * D_DIM, nullptr, q, G_NUM, D_DIM, H_DIM);
        gemm(tmp,   wtT + (size_t)i * H_DIM * D_DIM, cinB + (size_t)i * H_DIM, u, N, D_DIM, H_DIM);
        add_gather<<<g1d((long long)N * H_DIM), 256, 0, stream>>>(u, q, batch, N);
        gemm(u, chT + (size_t)i * H_DIM * H_DIM, chB + (size_t)i * H_DIM, v, N, H_DIM, H_DIM);
        gemm(v, coT + (size_t)i * D_DIM * H_DIM, coB + (size_t)i * D_DIM, tmp, N, H_DIM, D_DIM);
        mul_ew<<<g1d((long long)N * D_DIM), 256, 0, stream>>>(tmp, x, xcur, N * D_DIM);
    }

    // --- final 3-layer GCN (embed = n2) ---
    conv(xcur, w0t, b0, nullptr, n0);
    conv(n0,   w1t, b1, n0,      n1);
    conv(n1,   w2t, b2, n1,      n2);

    // --- GPFplusAtt ---
    att_weights<<<g1d(N), 256, 0, stream>>>(n2, aW, aB, wbuf, N);
    prompt_add<<<g1d((long long)N * D_DIM), 256, 0, stream>>>(n2, wbuf, plist, N * D_DIM);

    // --- readout: graph means -> idx gather -> class prototypes -> cosine softmax ---
    segmean(n2, m0);
    gather_rows<<<g1d((long long)NI * D_DIM), 256, 0, stream>>>(m0, idx, raw, NI * D_DIM);
    fill_kernel<<<g1d(C_NUM * D_DIM), 256, 0, stream>>>(csum, 0.0f, C_NUM * D_DIM);
    fill_kernel<<<g1d(C_NUM), 256, 0, stream>>>(ccnt, 0.0f, C_NUM);
    label_count<<<g1d(NI), 256, 0, stream>>>(labels, ccnt, NI);
    class_sum<<<g1d((long long)NI * D_DIM), 256, 0, stream>>>(raw, labels, csum, NI * D_DIM);
    class_norm<<<C_NUM, D_DIM, 0, stream>>>(csum, ccnt, an);
    final_softmax<<<NI, D_DIM, 0, stream>>>(raw, an, out);
}
